// CudaTensorProduct_65609920413716
// MI455X (gfx1250) — matrix operand order repaired and checked by reading, not compiled, not run on hardware
//
#include <hip/hip_runtime.h>
#include <hip/hip_bf16.h>
#include <stdint.h>

// CG tensor product as bf16 WMMA GEMM:  out[B,324] = kron(in1,in2)[B,324] @ W^T
//   B=65536, K = 324 -> pad 352 (11 x 32), N = 324 -> 352 cols alloc (11 chunks x 32)
// 97 MB HBM traffic -> 4.2 us floor @ 23.3 TB/s; 13.8 GFLOP dense bf16 WMMA ~3 us -> HBM-bound.
// W stored K-major, stride 360 halfs (180 dwords) => conflict-free half-wave ds_load_b128;
// each 32-column N-chunk is one contiguous 23,040 B block, staged global->LDS with async
// (ASYNCcnt) DMA, double-buffered so staging of chunk c+1 overlaps the 22 WMMAs of chunk c.
// sched_group_barrier pins a DS(4) + 22x[WMMA(1),DS(2)] issue order so B-fragment loads run
// one step ahead of their consuming WMMA (s_wait_dscnt 2 instead of 0).

typedef __attribute__((ext_vector_type(16))) __bf16 v16bf;
typedef __attribute__((ext_vector_type(8)))  __bf16 v8bf;
typedef __attribute__((ext_vector_type(8)))  float  v8f;

#define DIM1 36
#define DIM2 9
#define NOUT 324
#define KSTEPS 11          // K = 11 * 32 = 352 (>= 324)
#define KSTRIDE 360        // padded K stride in halfs (global AND LDS layout)
#define NT_CHUNK 2         // N-tiles per LDS chunk
#define NCHUNKS 11         // 22 tile slots; slot 21 is zero padding
#define CHUNK_COLS 32      // NT_CHUNK * 16
#define NPAD_ALLOC 352     // W columns allocated+zeroed = NCHUNKS * CHUNK_COLS
#define CHUNK_BYTES (CHUNK_COLS * KSTRIDE * 2)   // 23,040 B, contiguous in global W
#define BLOCK 128

__global__ void zero_w_kernel(uint32_t* w, int n) {
  int i = blockIdx.x * blockDim.x + threadIdx.x;
  if (i < n) w[i] = 0u;
}

__global__ void scatter_w_kernel(const int* __restrict__ i1, const int* __restrict__ i2,
                                 const int* __restrict__ io, const float* __restrict__ cb,
                                 int nnz, __bf16* __restrict__ W) {
  int t = blockIdx.x * blockDim.x + threadIdx.x;
  if (t < nnz) {
    int k = i1[t] * DIM2 + i2[t];        // kron index (K dim)
    int n = io[t];                       // output index (N dim)
    W[n * KSTRIDE + k] = (__bf16)cb[t];  // column-major [N][K], K-stride 360
  }
}

__global__ __launch_bounds__(BLOCK)
void tp_wmma_kernel(const float* __restrict__ in1, const float* __restrict__ in2,
                    const __bf16* __restrict__ W, float* __restrict__ out, int batch) {
  __shared__ __align__(16) __bf16 Wlds[2][CHUNK_COLS * KSTRIDE];   // 2 x 23,040 B

  const int tid  = threadIdx.x;
  const int lane = tid & 31;
  const int wave = tid >> 5;
  const bool hib = (lane >= 16);
  const int colHalf = (lane & 15);
  const int rowBase = (blockIdx.x * 4 + wave) * 16;
  const int b = rowBase + colHalf;
  const bool fullTile = (rowBase + 16 <= batch);

  // Async global->LDS stage of one contiguous chunk (ASYNCcnt, no VGPR round-trip).
  // Flat-aperture rule: low 32 bits of a generic LDS pointer are the LDS byte address.
  auto stage = [&](int c, int buf) {
    const uint32_t lbase = (uint32_t)(size_t)(&Wlds[buf][0]);
    const unsigned long long gbase =
        (unsigned long long)((const char*)W + (size_t)c * CHUNK_BYTES);
    for (int i = tid; i < CHUNK_BYTES / 16; i += BLOCK) {
      const uint32_t la = lbase + (uint32_t)i * 16u;
      const unsigned long long ga = gbase + (unsigned long long)i * 16ull;
      asm volatile("global_load_async_to_lds_b128 %0, %1, off"
                   :: "v"(la), "v"(ga) : "memory");
    }
  };

  stage(0, 0);   // prologue DMA overlaps the A-fragment build below

  // Per-lane input row -> registers (constant-indexed only).
  float r1[DIM1], r2[DIM2];
#pragma unroll
  for (int i = 0; i < DIM1; ++i) r1[i] = (b < batch) ? in1[b * DIM1 + i] : 0.f;
#pragma unroll
  for (int i = 0; i < DIM2; ++i) r2[i] = (b < batch) ? in2[b * DIM2 + i] : 0.f;

  // Build all 11 A-fragments once (reused across every N chunk/tile).
  // A 16x32, contiguous K order (same as B): lo half K = h (0..15); hi half: +16.
  // Both candidates are compile-time-indexed products; select per lane half (v_cndmask).
  v16bf afrag[KSTEPS];
#pragma unroll
  for (int s = 0; s < KSTEPS; ++s) {
#pragma unroll
    for (int h = 0; h < 16; ++h) {
      const int klo = s * 32 + h;
      const int khi = klo + 16;
      const float vlo = (klo < NOUT) ? r1[klo / DIM2] * r2[klo % DIM2] : 0.f;
      const float vhi = (khi < NOUT) ? r1[khi / DIM2] * r2[khi % DIM2] : 0.f;
      afrag[s][h] = (__bf16)(hib ? vhi : vlo);
    }
  }

  for (int c = 0; c < NCHUNKS; ++c) {
    const int cur = c & 1;
    // My stage(c) DMA done; barrier also proves every wave finished its ds reads of
    // buffer cur^1 (compiler waits dscnt before barrier) and its own stage(c) wait.
    asm volatile("s_wait_asynccnt 0x0" ::: "memory");
    __syncthreads();
    if (c + 1 < NCHUNKS) stage(c + 1, cur ^ 1);   // overlaps the compute below

    const __bf16* wbuf = &Wlds[cur][0];
    v8f acc[NT_CHUNK];
#pragma unroll
    for (int t = 0; t < NT_CHUNK; ++t) acc[t] = (v8f){};

    auto loadB = [&](int t, int s) -> v16bf {
      // B 32x16, contiguous K order (same as A): lane holds column (lane&15); lo lanes K=s*32+0..15, hi lanes +16.
      const __bf16* bp = &wbuf[(t * 16 + colHalf) * KSTRIDE + s * 32 + (hib ? 16 : 0)];
      const v8bf blo = *(const v8bf*)bp;
      const v8bf bhi = *(const v8bf*)(bp + 8);
      return __builtin_shufflevector(blo, bhi,
             0, 1, 2, 3, 4, 5, 6, 7, 8, 9, 10, 11, 12, 13, 14, 15);
    };

    // Depth-1 prefetch across the flattened (tile, k-step) pipeline.
    v16bf bcur = loadB(0, 0);
#pragma unroll
    for (int it = 0; it < NT_CHUNK * KSTEPS; ++it) {
      const int t = it / KSTEPS;
      const int s = it % KSTEPS;
      v16bf bnext = bcur;
      if (it + 1 < NT_CHUNK * KSTEPS)
        bnext = loadB((it + 1) / KSTEPS, (it + 1) % KSTEPS);
      acc[t] = __builtin_amdgcn_wmma_f32_16x16x32_bf16(
          /*neg_a=*/false, afrag[s], /*neg_b=*/false, bcur,
          /*c_mod=*/(short)0, acc[t], /*reuse_a=*/false, /*reuse_b=*/false);
      bcur = bnext;
    }

#if __has_builtin(__builtin_amdgcn_sched_group_barrier)
    // Pin issue order: 4 DS-reads up front, then [1 WMMA, 2 DS-reads] repeating, so each
    // WMMA's B fragment was loaded one step earlier (s_wait_dscnt 2, loads overlap WMMA).
    __builtin_amdgcn_sched_group_barrier(0x0100, 4, 0);   // DS read x4 (frags 0,1)
#pragma unroll
    for (int it = 0; it < NT_CHUNK * KSTEPS; ++it) {
      __builtin_amdgcn_sched_group_barrier(0x0008, 1, 0); // 1 WMMA
      if (it + 1 < NT_CHUNK * KSTEPS)
        __builtin_amdgcn_sched_group_barrier(0x0100, 2, 0); // next B fragment
    }
#endif

    // D layout: VGPR r -> M = r (lo lanes) / r+8 (hi lanes), N = lane&15.
#pragma unroll
    for (int t = 0; t < NT_CHUNK; ++t) {
      const int n = c * CHUNK_COLS + t * 16 + colHalf;
      if (n < NOUT) {
        if (fullTile) {
          float* op = out + (size_t)(rowBase + (hib ? 8 : 0)) * NOUT + n;
#pragma unroll
          for (int r = 0; r < 8; ++r) op[(size_t)r * NOUT] = acc[t][r];
        } else {
#pragma unroll
          for (int r = 0; r < 8; ++r) {
            const int row = rowBase + r + (hib ? 8 : 0);
            if (row < batch) out[(size_t)row * NOUT + n] = acc[t][r];
          }
        }
      }
    }
  }
}

extern "C" void kernel_launch(void* const* d_in, const int* in_sizes, int n_in,
                              void* d_out, int out_size, void* d_ws, size_t ws_size,
                              hipStream_t stream) {
  (void)n_in; (void)out_size; (void)ws_size;
  const float* in1 = (const float*)d_in[0];
  const float* in2 = (const float*)d_in[1];
  const int*   i1  = (const int*)d_in[2];
  const int*   i2  = (const int*)d_in[3];
  const int*   io  = (const int*)d_in[4];
  const float* cb  = (const float*)d_in[5];
  const int nnz   = in_sizes[5];
  const int batch = in_sizes[0] / DIM1;

  __bf16* W = (__bf16*)d_ws;                        // 352*360*2 = 253,440 B of workspace
  const int wWords = (NPAD_ALLOC * KSTRIDE * 2) / 4;

  zero_w_kernel<<<(wWords + 255) / 256, 256, 0, stream>>>((uint32_t*)d_ws, wWords);
  scatter_w_kernel<<<(nnz + 255) / 256, 256, 0, stream>>>(i1, i2, io, cb, nnz, W);

  dim3 grid((batch + 63) / 64);                     // 64 batch rows per workgroup (4 waves)
  tp_wmma_kernel<<<grid, BLOCK, 0, stream>>>(in1, in2, W, (float*)d_out, batch);
}
